// GraspSamplerVAE_61263413510334
// MI455X (gfx1250) — matrix operand order repaired and checked by reading, not compiled, not run on hardware
//
#include <hip/hip_runtime.h>

typedef _Float16 h16;
typedef __attribute__((ext_vector_type(16))) _Float16 v16h;
typedef __attribute__((ext_vector_type(8)))  _Float16 v8h;
typedef __attribute__((ext_vector_type(8)))  float    v8f;
typedef __attribute__((ext_vector_type(4)))  int      v4i;

#define BN_EPS 1e-5f

static inline unsigned cdiv(size_t a, size_t b){ return (unsigned)((a + b - 1) / b); }

// ---------------------------------------------------------------------------
// Async global->LDS staging (gfx1250 GLOBAL_LOAD_ASYNC_TO_LDS_B128, ASYNCcnt)
// Builtin signature: (v4i AS1* src_global, v4i AS3* dst_lds, imm off, imm cpol)
// Guarded: falls back to synchronous LDS stores if the builtin is absent.
// ---------------------------------------------------------------------------
#if __has_builtin(__builtin_amdgcn_global_load_async_to_lds_b128)
#define HAS_ASYNC_LDS 1
__device__ inline void async_copy_b128(const h16* g, h16* l){
    __builtin_amdgcn_global_load_async_to_lds_b128(
        (__attribute__((address_space(1))) v4i*)g,
        (__attribute__((address_space(3))) v4i*)l, 0, 0);
}
#else
__device__ inline void async_copy_b128(const h16* g, h16* l){
    *(v8h*)l = *(const v8h*)g;          // synchronous fallback
}
#endif

__device__ inline void wait_async(){
#if defined(HAS_ASYNC_LDS)
#if __has_builtin(__builtin_amdgcn_s_wait_asynccnt)
    __builtin_amdgcn_s_wait_asynccnt(0);
#else
    asm volatile("s_wait_asynccnt 0x0" ::: "memory");
#endif
#endif
}

// ---------------------------------------------------------------------------
// WMMA fragment loaders (CDNA5 16x16x32 f16, wave32; layouts per ISA 7.12.2)
// ---------------------------------------------------------------------------
// A (16x32, MxK): lanes 0-15 -> M=lane, K chunks [0..7]+[16..23];
//                 lanes 16-31 -> M=lane-16, K chunks [8..15]+[24..31].
__device__ inline v16h frag_a(const h16* __restrict__ A, int ldk, int row0, int k0, int lane){
    int m  = lane & 15;
    int hi = lane >> 4;                       // 0 or 1
    const h16* p = A + (size_t)(row0 + m) * ldk + k0 + hi * 8;
    v8h a0 = *(const v8h*)(p);
    v8h a1 = *(const v8h*)(p + 16);
    return __builtin_shufflevector(a0, a1, 0,1,2,3,4,5,6,7,8,9,10,11,12,13,14,15);
}
// B fragment from LDS panel lb[64][32] (row = output channel within the
// 64-wide N block, 32 K-halfs contiguous): lane -> N=lane&15,
// lanes 0-15 hold K=[0..7]+[16..23], lanes 16-31 hold K=[8..15]+[24..31].
__device__ inline v16h frag_b_lds(const h16* lb, int j, int lane){
    int n  = lane & 15;
    int hi = lane >> 4;
    const h16* p = lb + ((j * 16 + n) * 32 + hi * 8);
    v8h b0 = *(const v8h*)(p);
    v8h b1 = *(const v8h*)(p + 16);
    return __builtin_shufflevector(b0, b1, 0,1,2,3,4,5,6,7,8,9,10,11,12,13,14,15);
}

// ---------------------------------------------------------------------------
// GEMM: C[M,N] = A[M,Kp] (f16) * W[N,Kp]^T (f16) + bias[N], f32 accumulate.
// grid = (ceil(M/16/4), N/64), block = 128 (4 waves).
// Block stages the shared 64-col weight panel into LDS (async, double
// buffered, 32 K per step); each wave computes a 16x64 strip = 4 WMMAs/step.
// ---------------------------------------------------------------------------
__global__ __launch_bounds__(128) void k_gemm_wmma(
    const h16* __restrict__ A, const h16* __restrict__ W,
    const float* __restrict__ bias, float* __restrict__ C,
    int M, int N, int Kp)
{
    __shared__ __align__(16) h16 lb[2][64 * 32];   // 2 x 4KB double buffer
    int lane = threadIdx.x & 31;
    int wave = threadIdx.x >> 5;
    int tid  = threadIdx.x;
    int tiles_m = M >> 4;
    int tm   = blockIdx.x * 4 + wave;
    bool active = tm < tiles_m;               // wave-uniform
    int row0 = tm << 4;
    int col0 = blockIdx.y << 6;               // 64 columns per block
    int steps = Kp >> 5;

    auto stage = [&](int s, int b){
        // panel = W[col0..col0+63][32s..32s+31] : 64 rows x 64B = 256 x b128
        #pragma unroll
        for (int e = tid; e < 256; e += 128){
            int r   = e >> 2;
            int sub = e & 3;
            const h16* g = W + (size_t)(col0 + r) * Kp + (s << 5) + sub * 8;
            async_copy_b128(g, &lb[b][r * 32 + sub * 8]);
        }
    };

    stage(0, 0);
    v8f acc[4] = {};
    for (int s = 0; s < steps; ++s){
        wait_async();                          // own stage(s) transfers complete
        __syncthreads();                       // everyone's stage(s) visible; buf s+1 free
        if (s + 1 < steps) stage(s + 1, (s + 1) & 1);
        if (active){
            int k0 = s << 5;
            v16h a = frag_a(A, Kp, row0, k0, lane);
            const h16* panel = &lb[s & 1][0];
            #pragma unroll
            for (int j = 0; j < 4; ++j){
                v16h bf = frag_b_lds(panel, j, lane);
                acc[j] = __builtin_amdgcn_wmma_f32_16x16x32_f16(
                             false, a, false, bf, (short)0, acc[j], false, false);
            }
        }
    }
    if (active){
        int n  = lane & 15;
        int mb = (lane >> 4) * 8;
        #pragma unroll
        for (int j = 0; j < 4; ++j){
            int col = col0 + j * 16 + n;
            float bv = bias ? bias[col] : 0.f;
            #pragma unroll
            for (int r = 0; r < 8; ++r)
                C[(size_t)(row0 + mb + r) * N + col] = acc[j][r] + bv;
        }
    }
}

// ---------------------------------------------------------------------------
// Farthest point sampling: one block per batch, seeded at index 0.
// ---------------------------------------------------------------------------
__global__ void k_fps(const float* __restrict__ xyz, int N, int S, int* __restrict__ out){
    int b = blockIdx.x;
    const float* X = xyz + (size_t)b * N * 3;
    __shared__ float dist[1024];
    __shared__ float rv[256];
    __shared__ int   ri[256];
    __shared__ int   lastS;
    for (int i = threadIdx.x; i < N; i += blockDim.x) dist[i] = 1e10f;
    if (threadIdx.x == 0){ lastS = 0; out[b * S] = 0; }
    __syncthreads();
    for (int it = 1; it < S; ++it){
        int last = lastS;
        float lx = X[last*3], ly = X[last*3+1], lz = X[last*3+2];
        float best = -1.f; int bi = 0;
        for (int i = threadIdx.x; i < N; i += blockDim.x){
            float dx = X[i*3]-lx, dy = X[i*3+1]-ly, dz = X[i*3+2]-lz;
            float nd = fminf(dist[i], dx*dx + dy*dy + dz*dz);
            dist[i] = nd;
            if (nd > best){ best = nd; bi = i; }
        }
        rv[threadIdx.x] = best; ri[threadIdx.x] = bi;
        __syncthreads();
        for (int o = blockDim.x >> 1; o > 0; o >>= 1){
            if ((int)threadIdx.x < o && rv[threadIdx.x + o] > rv[threadIdx.x]){
                rv[threadIdx.x] = rv[threadIdx.x + o];
                ri[threadIdx.x] = ri[threadIdx.x + o];
            }
            __syncthreads();
        }
        if (threadIdx.x == 0){ lastS = ri[0]; out[b * S + it] = ri[0]; }
        __syncthreads();
    }
}

__global__ void k_gather_centers(const float* __restrict__ xyz, const int* __restrict__ cidx,
                                 float* __restrict__ cent, int N, int S, int total){
    int i = blockIdx.x * blockDim.x + threadIdx.x;   // over B*S*3
    if (i >= total) return;
    int c = i % 3, s = (i / 3) % S, b = i / (3 * S);
    cent[i] = xyz[((size_t)b * N + cidx[b * S + s]) * 3 + c];
}

// ---------------------------------------------------------------------------
// Ball query + group + center + concat + f16-convert with K zero-padding.
// One wave per (b,s). Output rows ((b*S+s)*ns + j) of width Kp.
// ---------------------------------------------------------------------------
__global__ void k_ball_group(const float* __restrict__ xyz, const float* __restrict__ cent,
                             const float* __restrict__ feats, h16* __restrict__ A,
                             int N, int S, int ns, int C, int Kp, float r2)
{
    int bs = blockIdx.x;          // b*S + s
    int b  = bs / S;
    int lane = threadIdx.x;
    __shared__ int slot[128];
    float cx = cent[(size_t)bs*3], cy = cent[(size_t)bs*3+1], cz = cent[(size_t)bs*3+2];
    const float* X = xyz + (size_t)b * N * 3;
    int count = 0;
    for (int base = 0; base < N && count < ns; base += 32){
        int i = base + lane;
        bool in = false;
        if (i < N){
            float dx = X[i*3]-cx, dy = X[i*3+1]-cy, dz = X[i*3+2]-cz;
            in = (dx*dx + dy*dy + dz*dz) < r2;
        }
        unsigned m = (unsigned)__ballot(in);
        int pos = count + __popc(m & ((1u << lane) - 1u));
        if (in && pos < ns) slot[pos] = i;
        count += __popc(m);
    }
    if (count > ns) count = ns;
    __syncthreads();
    int fb = (count > 0) ? slot[0] : 0;
    const float* F = feats + (size_t)b * N * C;
    for (int j = 0; j < ns; ++j){
        int idx = (j < count) ? slot[j] : fb;
        h16* row = A + ((size_t)bs * ns + j) * Kp;
        for (int c = lane; c < Kp; c += 32){
            float v;
            if (c < 3)            v = X[idx*3 + c] - cent[(size_t)bs*3 + c];
            else if (c < 3 + C)   v = F[(size_t)idx * C + (c - 3)];
            else                  v = 0.f;
            row[c] = (h16)v;
        }
    }
}

// sa_all grouping: rows (b*S+s), channels [xyz | feats | pad]
__global__ void k_sa3_build(const float* __restrict__ xyz, const float* __restrict__ feats,
                            h16* __restrict__ A, int BS, int C, int Kp){
    size_t i = (size_t)blockIdx.x * blockDim.x + threadIdx.x;
    if (i >= (size_t)BS * Kp) return;
    int c = (int)(i % Kp);
    size_t bs = i / Kp;
    float v = 0.f;
    if (c < 3)          v = xyz[bs*3 + c];
    else if (c < 3 + C) v = feats[bs*C + (c - 3)];
    A[i] = (h16)v;
}

// ---------------------------------------------------------------------------
// Train-mode batchnorm (biased variance) + ReLU
// ---------------------------------------------------------------------------
__global__ void k_bn_stats(const float* __restrict__ X, int M, int N,
                           float* __restrict__ mean, float* __restrict__ var){
    int c = blockIdx.x;
    float s = 0.f, sq = 0.f;
    for (int i = threadIdx.x; i < M; i += 256){
        float v = X[(size_t)i * N + c];
        s += v; sq += v * v;
    }
    __shared__ float ss[256], sv[256];
    ss[threadIdx.x] = s; sv[threadIdx.x] = sq;
    __syncthreads();
    for (int o = 128; o > 0; o >>= 1){
        if ((int)threadIdx.x < o){ ss[threadIdx.x] += ss[threadIdx.x+o]; sv[threadIdx.x] += sv[threadIdx.x+o]; }
        __syncthreads();
    }
    if (threadIdx.x == 0){
        float m = ss[0] / (float)M;
        mean[c] = m;
        var[c]  = sv[0] / (float)M - m * m;
    }
}

__global__ void k_bn_relu(const float* __restrict__ X, const float* __restrict__ mean,
                          const float* __restrict__ var, const float* __restrict__ g,
                          const float* __restrict__ be, h16* __restrict__ Yh,
                          float* __restrict__ Yf, int M, int N){
    size_t i = (size_t)blockIdx.x * blockDim.x + threadIdx.x;
    if (i >= (size_t)M * N) return;
    int c = (int)(i % N);
    float y = g[c] * (X[i] - mean[c]) * rsqrtf(var[c] + BN_EPS) + be[c];
    y = fmaxf(y, 0.f);
    if (Yh) Yh[i] = (h16)y;
    if (Yf) Yf[i] = y;
}

// max over the ns dimension: H f16 [BS, ns, C] -> out f32 [BS, C]
__global__ void k_maxpool(const h16* __restrict__ H, float* __restrict__ out,
                          int BS, int ns, int C){
    int i = blockIdx.x * blockDim.x + threadIdx.x;
    if (i >= BS * C) return;
    int c = i % C, bs = i / C;
    const h16* p = H + (size_t)bs * ns * C + c;
    float m = -1e30f;
    for (int j = 0; j < ns; ++j) m = fmaxf(m, (float)p[(size_t)j * C]);
    out[i] = m;
}

// f32 -> f16 copy with K zero-pad: out [R, Kp], in [R, K]
__global__ void k_cvt_w(const float* __restrict__ W, h16* __restrict__ O,
                        int R, int K, int Kp){
    int i = blockIdx.x * blockDim.x + threadIdx.x;
    if (i >= R * Kp) return;
    int k = i % Kp, r = i / Kp;
    O[i] = (h16)((k < K) ? W[(size_t)r * K + k] : 0.f);
}

// feats = concat(pc, broadcast(vec)) : out [B,N,3+D]
__global__ void k_feats_cat(const float* __restrict__ pc, const float* __restrict__ vec,
                            float* __restrict__ out, int B, int N, int D){
    int C = 3 + D;
    size_t i = (size_t)blockIdx.x * blockDim.x + threadIdx.x;
    if (i >= (size_t)B * N * C) return;
    int c = (int)(i % C);
    size_t bn = i / C;
    int b = (int)(bn / N);
    out[i] = (c < 3) ? pc[bn * 3 + c] : vec[b * D + (c - 3)];
}

// small head matvec: out[r,o] = dot(X[r,:K], W[o,:K]) + b[o]
__global__ void k_matvec(const float* __restrict__ X, const float* __restrict__ W,
                         const float* __restrict__ bias, float* __restrict__ out,
                         int O, int K){
    int t = blockIdx.x;
    int r = t / O, o = t % O;
    float s = 0.f;
    for (int k = threadIdx.x; k < K; k += 256)
        s += X[(size_t)r * K + k] * W[(size_t)o * K + k];
    __shared__ float ss[256];
    ss[threadIdx.x] = s;
    __syncthreads();
    for (int off = 128; off > 0; off >>= 1){
        if ((int)threadIdx.x < off) ss[threadIdx.x] += ss[threadIdx.x + off];
        __syncthreads();
    }
    if (threadIdx.x == 0) out[t] = ss[0] + bias[o];
}

__device__ inline unsigned pcg(unsigned v){
    unsigned s = v * 747796405u + 2891336453u;
    unsigned w = ((s >> ((s >> 28u) + 4u)) ^ s) * 277803737u;
    return (w >> 22u) ^ w;
}
// deterministic reparameterization (hash -> Box-Muller)
__global__ void k_z(const float* __restrict__ mu, const float* __restrict__ logvar,
                    float* __restrict__ z, int n){
    int i = blockIdx.x * blockDim.x + threadIdx.x;
    if (i >= n) return;
    float u1 = (float)(pcg(2u*i) + 1u) * 2.3283064e-10f;
    float u2 = (float)pcg(2u*i + 1u) * 2.3283064e-10f;
    float eps = sqrtf(-2.f * logf(u1)) * cosf(6.28318530718f * u2);
    z[i] = mu[i] + eps * expf(0.5f * logvar[i]);
}

// pack outputs: [qt (B*7) | conf (B) | mu (B*2) | logvar (B*2)]
__global__ void k_final(const float* __restrict__ q, const float* __restrict__ t,
                        const float* __restrict__ conf, const float* __restrict__ mu,
                        const float* __restrict__ logvar, float* __restrict__ out, int B){
    int i = blockIdx.x * blockDim.x + threadIdx.x;
    if (i >= B) return;
    float q0 = q[i*4], q1 = q[i*4+1], q2 = q[i*4+2], q3 = q[i*4+3];
    float nrm = fmaxf(sqrtf(q0*q0 + q1*q1 + q2*q2 + q3*q3), 1e-12f);
    out[i*7+0] = q0/nrm; out[i*7+1] = q1/nrm; out[i*7+2] = q2/nrm; out[i*7+3] = q3/nrm;
    out[i*7+4] = t[i*3]; out[i*7+5] = t[i*3+1]; out[i*7+6] = t[i*3+2];
    out[7*B + i] = 1.f / (1.f + expf(-conf[i]));
    out[8*B + i*2]     = mu[i*2];     out[8*B + i*2 + 1]     = mu[i*2 + 1];
    out[10*B + i*2]    = logvar[i*2]; out[10*B + i*2 + 1]    = logvar[i*2 + 1];
}

// ---------------------------------------------------------------------------
// Host orchestration
// ---------------------------------------------------------------------------
extern "C" void kernel_launch(void* const* d_in, const int* in_sizes, int n_in,
                              void* d_out, int out_size, void* d_ws, size_t ws_size,
                              hipStream_t stream)
{
    (void)in_sizes; (void)n_in; (void)out_size; (void)ws_size;
    const int B = 32, N = 1024, GD = 16, LAT = 2;
    auto F = [&](int i){ return (const float*)d_in[i]; };
    const float* pc    = F(0);
    const float* grasp = F(1);
    // jax pytree flatten order (dict keys sorted; tuples in order W,b,g,be):
    // 2,3: conf | 4..11: dec.fc | 12..23: dec.sa1 | 24..35: dec.sa2 | 36..47: dec.sa3
    // 48..55: enc.fc | 56..67: enc.sa1 | 68..79: enc.sa2 | 80..91: enc.sa3
    // 92,93: logvar | 94,95: mu | 96,97: q | 98,99: t

    char* wsb = (char*)d_ws;
    size_t off = 0;
    auto alloc = [&](size_t bytes)->void*{
        void* p = wsb + off;
        off = (off + bytes + 255) & ~(size_t)255;
        return p;
    };
    int*   cidx1  = (int*)  alloc((size_t)B*128*4);
    int*   cidx2  = (int*)  alloc((size_t)B*32*4);
    float* cent1  = (float*)alloc((size_t)B*128*3*4);
    float* cent2  = (float*)alloc((size_t)B*32*3*4);
    float* feats0 = (float*)alloc((size_t)B*N*19*4);
    float* pool1  = (float*)alloc((size_t)B*128*128*4);
    float* pool2  = (float*)alloc((size_t)B*32*256*4);
    float* gvec   = (float*)alloc((size_t)B*512*4);
    float* y_enc  = (float*)alloc((size_t)B*1024*4);
    float* y_dec  = (float*)alloc((size_t)B*1024*4);
    float* mu     = (float*)alloc((size_t)B*LAT*4);
    float* logv   = (float*)alloc((size_t)B*LAT*4);
    float* zbuf   = (float*)alloc((size_t)B*LAT*4);
    float* qh     = (float*)alloc((size_t)B*4*4);
    float* th     = (float*)alloc((size_t)B*3*4);
    float* ch     = (float*)alloc((size_t)B*1*4);
    float* meanb  = (float*)alloc(1024*4);
    float* varb   = (float*)alloc(1024*4);
    h16*   W16    = (h16*)  alloc((size_t)1024*1024*2);
    float* G      = (float*)alloc((size_t)262144*128*4);   // gemm f32 out (max)
    h16*   Hp     = (h16*)  alloc((size_t)262144*128*2);   // f16 act ping
    h16*   Hq     = (h16*)  alloc((size_t)262144*128*2);   // f16 act pong

    auto run_layer = [&](const h16* Ain, int M, int Kp, int wi, int Cin, int Cout,
                         h16* Hout, float* Fout){
        k_cvt_w<<<cdiv((size_t)Cout*Kp,256),256,0,stream>>>(F(wi), W16, Cout, Cin, Kp);
        int tiles_m = M / 16;
        dim3 grid(cdiv(tiles_m, 4), Cout / 64);
        k_gemm_wmma<<<grid,128,0,stream>>>(Ain, W16, F(wi+1), G, M, Cout, Kp);
        k_bn_stats<<<Cout,256,0,stream>>>(G, M, Cout, meanb, varb);
        k_bn_relu<<<cdiv((size_t)M*Cout,256),256,0,stream>>>(G, meanb, varb,
                                                             F(wi+2), F(wi+3),
                                                             Hout, Fout, M, Cout);
    };

    auto run_pointnet = [&](const float* featsIn, int C0,
                            int sa1, int sa2, int sa3, int fci, float* yout){
        // ---- SA1: npoint=128, radius=0.02, nsample=64 ----
        k_fps<<<B,256,0,stream>>>(pc, N, 128, cidx1);
        k_gather_centers<<<cdiv((size_t)B*128*3,256),256,0,stream>>>(pc, cidx1, cent1, N, 128, B*128*3);
        int Kp1 = ((3 + C0) + 31) / 32 * 32;
        k_ball_group<<<B*128,32,0,stream>>>(pc, cent1, featsIn, Hp, N, 128, 64, C0, Kp1, 0.02f*0.02f);
        int M1 = B*128*64;
        run_layer(Hp, M1, Kp1, sa1+0, 3+C0, 64,  Hq, nullptr);
        run_layer(Hq, M1, 64,  sa1+4, 64,   64,  Hp, nullptr);
        run_layer(Hp, M1, 64,  sa1+8, 64,   128, Hq, nullptr);
        k_maxpool<<<cdiv((size_t)B*128*128,256),256,0,stream>>>(Hq, pool1, B*128, 64, 128);

        // ---- SA2: npoint=32, radius=0.04, nsample=128 ----
        k_fps<<<B,256,0,stream>>>(cent1, 128, 32, cidx2);
        k_gather_centers<<<cdiv((size_t)B*32*3,256),256,0,stream>>>(cent1, cidx2, cent2, 128, 32, B*32*3);
        k_ball_group<<<B*32,32,0,stream>>>(cent1, cent2, pool1, Hp, 128, 32, 128, 128, 160, 0.04f*0.04f);
        int M2 = B*32*128;
        run_layer(Hp, M2, 160, sa2+0, 131, 128, Hq, nullptr);
        run_layer(Hq, M2, 128, sa2+4, 128, 128, Hp, nullptr);
        run_layer(Hp, M2, 128, sa2+8, 128, 256, Hq, nullptr);
        k_maxpool<<<cdiv((size_t)B*32*256,256),256,0,stream>>>(Hq, pool2, B*32, 128, 256);

        // ---- SA3 (group-all over S=32) ----
        k_sa3_build<<<cdiv((size_t)B*32*288,256),256,0,stream>>>(cent2, pool2, Hp, B*32, 256, 288);
        int M3 = B*32;
        run_layer(Hp, M3, 288, sa3+0, 259, 256, Hq, nullptr);
        run_layer(Hq, M3, 256, sa3+4, 256, 256, Hp, nullptr);
        run_layer(Hp, M3, 256, sa3+8, 256, 512, Hq, nullptr);
        k_maxpool<<<cdiv((size_t)B*512,256),256,0,stream>>>(Hq, gvec, B, 32, 512);

        // ---- FC stack [512 -> 1024 -> 1024] with batch BN ----
        k_cvt_w<<<cdiv((size_t)B*512,256),256,0,stream>>>(gvec, Hp, B, 512, 512);
        run_layer(Hp, B, 512,  fci+0, 512,  1024, Hq, nullptr);
        run_layer(Hq, B, 1024, fci+4, 1024, 1024, Hp, yout);
    };

    // ===== Encoder =====
    k_feats_cat<<<cdiv((size_t)B*N*(3+GD),256),256,0,stream>>>(pc, grasp, feats0, B, N, GD);
    run_pointnet(feats0, 3+GD, /*sa1*/56, /*sa2*/68, /*sa3*/80, /*fc*/48, y_enc);

    // mu / logvar / z
    k_matvec<<<B*LAT,256,0,stream>>>(y_enc, F(94), F(95), mu,   LAT, 1024);
    k_matvec<<<B*LAT,256,0,stream>>>(y_enc, F(92), F(93), logv, LAT, 1024);
    k_z<<<1,64,0,stream>>>(mu, logv, zbuf, B*LAT);

    // ===== Decoder =====
    k_feats_cat<<<cdiv((size_t)B*N*(3+LAT),256),256,0,stream>>>(pc, zbuf, feats0, B, N, LAT);
    run_pointnet(feats0, 3+LAT, /*sa1*/12, /*sa2*/24, /*sa3*/36, /*fc*/4, y_dec);

    // heads
    k_matvec<<<B*4,256,0,stream>>>(y_dec, F(96), F(97), qh, 4, 1024);
    k_matvec<<<B*3,256,0,stream>>>(y_dec, F(98), F(99), th, 3, 1024);
    k_matvec<<<B*1,256,0,stream>>>(y_dec, F(2),  F(3),  ch, 1, 1024);
    k_final<<<1,64,0,stream>>>(qh, th, ch, mu, logv, (float*)d_out, B);
}
